// G_Tensor3D_1460288881029
// MI455X (gfx1250) — compile-verified, matrix operand order checked by reading, not run on hardware
//
#include <hip/hip_runtime.h>

#define GXC 512
#define NFC 32
#define XDC 1024

typedef __attribute__((ext_vector_type(16))) _Float16 v16h;
typedef __attribute__((ext_vector_type(8)))  float    v8f;

#define TILES_PER_WAVE   8
#define WAVES_PER_BLOCK  8
#define THREADS          (WAVES_PER_BLOCK * 32)

// Pack one column of a 32x32 row-major weight matrix into the CDNA5 WMMA
// 16-bit B-matrix per-lane layout: element e holds K = (e<8 ? kb0+e : kb1+e-8).
__device__ __forceinline__ v16h pack_w_col(const float* __restrict__ W,
                                           int ncol, int kb0, int kb1) {
    v16h r;
#pragma unroll
    for (int e = 0; e < 8; ++e) r[e]     = (_Float16)W[(kb0 + e) * NFC + ncol];
#pragma unroll
    for (int e = 0; e < 8; ++e) r[8 + e] = (_Float16)W[(kb1 + e) * NFC + ncol];
    return r;
}

__global__ __launch_bounds__(THREADS)
void grid_mlp_kernel(const float* __restrict__ data,
                     const float* __restrict__ W1, const float* __restrict__ b1,
                     const float* __restrict__ W2, const float* __restrict__ b2,
                     const float* __restrict__ W3, const float* __restrict__ b3,
                     const int*   __restrict__ x0, const int* __restrict__ y0,
                     const int*   __restrict__ x1, const int* __restrict__ y1,
                     const float* __restrict__ lerp,
                     float*       __restrict__ out)
{
    __shared__ float hbuf_all[WAVES_PER_BLOCK * 16 * NFC];   // 16 KB total
    const int lane  = threadIdx.x & 31;
    const int wave  = threadIdx.x >> 5;
    float* hb = hbuf_all + wave * (16 * NFC);                // 16x32 f32 per wave

    const int half  = lane >> 4;       // which half-wave
    const int mrow  = lane & 15;       // A-matrix row owned by this lane
    const int kb0   = half ? 8  : 0;   // A/B layout K-groups for this lane
    const int kb1   = half ? 24 : 16;
    const int ncol  = lane & 15;       // C/D + B column owned by this lane
    const int mbase = half ? 8  : 0;   // C/D M offset for this half-wave

    // ---- per-wave setup: weights into B layout, biases broadcast ----------
    const v16h w1b0 = pack_w_col(W1, ncol,      kb0, kb1);
    const v16h w1b1 = pack_w_col(W1, ncol + 16, kb0, kb1);
    const v16h w2b0 = pack_w_col(W2, ncol,      kb0, kb1);
    const v16h w2b1 = pack_w_col(W2, ncol + 16, kb0, kb1);
    const float bias1a = b1[ncol], bias1b = b1[ncol + 16];
    const float bias2a = b2[ncol], bias2b = b2[ncol + 16];
    const float b3v = b3[0];

    const int wave_global = blockIdx.x * WAVES_PER_BLOCK + wave;

    for (int i = 0; i < TILES_PER_WAVE; ++i) {
        const int t = wave_global * TILES_PER_WAVE + i;   // 16-pixel tile id
        const int p = t * 16 + mrow;                      // pixel for this lane

        // ---- bilinear feature gather, produced directly in A layout ------
        const int xi0 = x0[p], yi0 = y0[p], xi1 = x1[p], yi1 = y1[p];
        const float w0  = lerp[2 * p + 0];
        const float w1v = lerp[2 * p + 1];
        const float wA = (1.f - w0) * (1.f - w1v);
        const float wB = w0 * (1.f - w1v);
        const float wC = (1.f - w0) * w1v;
        const float wD = w0 * w1v;
        const float* pa = data + ((size_t)yi0 * GXC + xi0) * NFC;
        const float* pb = data + ((size_t)yi0 * GXC + xi1) * NFC;
        const float* pc = data + ((size_t)yi1 * GXC + xi0) * NFC;
        const float* pd = data + ((size_t)yi1 * GXC + xi1) * NFC;

        v16h a;
#pragma unroll
        for (int g = 0; g < 2; ++g) {
            const int kb = g ? kb1 : kb0;
#pragma unroll
            for (int q = 0; q < 2; ++q) {
                const float4 va = *(const float4*)(pa + kb + 4 * q);  // 16B aligned
                const float4 vb = *(const float4*)(pb + kb + 4 * q);
                const float4 vc = *(const float4*)(pc + kb + 4 * q);
                const float4 vd = *(const float4*)(pd + kb + 4 * q);
                const int e = g * 8 + q * 4;
                a[e + 0] = (_Float16)(va.x * wA + vb.x * wB + vc.x * wC + vd.x * wD);
                a[e + 1] = (_Float16)(va.y * wA + vb.y * wB + vc.y * wC + vd.y * wD);
                a[e + 2] = (_Float16)(va.z * wA + vb.z * wB + vc.z * wC + vd.z * wD);
                a[e + 3] = (_Float16)(va.w * wA + vb.w * wB + vc.w * wC + vd.w * wD);
            }
        }

        // ---- layer 1: h = relu(feat @ W1 + b1), two 16x16x32 WMMAs -------
        v8f c0, c1;
#pragma unroll
        for (int r = 0; r < 8; ++r) { c0[r] = bias1a; c1[r] = bias1b; }
        c0 = __builtin_amdgcn_wmma_f32_16x16x32_f16(false, a, false, w1b0,
                                                    (short)0, c0, false, false);
        c1 = __builtin_amdgcn_wmma_f32_16x16x32_f16(false, a, false, w1b1,
                                                    (short)0, c1, false, false);

        // relu + C-layout -> LDS (16x32), then reload in A layout
#pragma unroll
        for (int r = 0; r < 8; ++r) {
            hb[(mbase + r) * NFC + ncol]      = fmaxf(c0[r], 0.f);
            hb[(mbase + r) * NFC + ncol + 16] = fmaxf(c1[r], 0.f);
        }
        v16h a2;
#pragma unroll
        for (int e = 0; e < 8; ++e) a2[e]     = (_Float16)hb[mrow * NFC + kb0 + e];
#pragma unroll
        for (int e = 0; e < 8; ++e) a2[8 + e] = (_Float16)hb[mrow * NFC + kb1 + e];

        // ---- layer 2 ------------------------------------------------------
#pragma unroll
        for (int r = 0; r < 8; ++r) { c0[r] = bias2a; c1[r] = bias2b; }
        c0 = __builtin_amdgcn_wmma_f32_16x16x32_f16(false, a2, false, w2b0,
                                                    (short)0, c0, false, false);
        c1 = __builtin_amdgcn_wmma_f32_16x16x32_f16(false, a2, false, w2b1,
                                                    (short)0, c1, false, false);
#pragma unroll
        for (int r = 0; r < 8; ++r) {
            hb[(mbase + r) * NFC + ncol]      = fmaxf(c0[r], 0.f);
            hb[(mbase + r) * NFC + ncol + 16] = fmaxf(c1[r], 0.f);
        }

        // ---- layer 3: 32->1, VALU dot on lanes 0..15 ----------------------
        if (lane < 16) {
            float s = b3v;
#pragma unroll
            for (int k = 0; k < NFC; ++k) s += hb[mrow * NFC + k] * W3[k];
            out[p] = s;
        }
    }
}

extern "C" void kernel_launch(void* const* d_in, const int* in_sizes, int n_in,
                              void* d_out, int out_size, void* d_ws, size_t ws_size,
                              hipStream_t stream) {
    // setup_inputs order: z, data, W1, b1, W2, b2, W3, b3, x0, y0, x1, y1, lerp
    const float* data = (const float*)d_in[1];
    const float* W1   = (const float*)d_in[2];
    const float* b1   = (const float*)d_in[3];
    const float* W2   = (const float*)d_in[4];
    const float* b2   = (const float*)d_in[5];
    const float* W3   = (const float*)d_in[6];
    const float* b3   = (const float*)d_in[7];
    const int*   x0   = (const int*)d_in[8];
    const int*   y0   = (const int*)d_in[9];
    const int*   x1   = (const int*)d_in[10];
    const int*   y1   = (const int*)d_in[11];
    const float* lw   = (const float*)d_in[12];
    float* out = (float*)d_out;

    const int tiles  = (XDC * XDC) / 16;                        // 65536
    const int blocks = tiles / (WAVES_PER_BLOCK * TILES_PER_WAVE); // 1024
    grid_mlp_kernel<<<blocks, THREADS, 0, stream>>>(
        data, W1, b1, W2, b2, W3, b3, x0, y0, x1, y1, lw, out);
}
